// pnet_2PointNet2_30270929502682
// MI455X (gfx1250) — matrix operand order repaired and checked by reading, not compiled, not run on hardware
//
#include <hip/hip_runtime.h>

// ---------------------------------------------------------------------------
// PointNet++ (semantic seg) forward for MI455X / gfx1250, wave32 + WMMA.
// All Dense+BN(+bias)+ReLU layers run as f16 WMMA GEMMs (v_wmma_f32_16x16x32_f16)
// with BN gamma folded into pre-packed B-layout weights and f32 accumulation.
// Each wave computes a 16 x (NT*16) output strip: the A (activation) fragment
// is loaded once per K-step and reused for NT WMMAs -> NT-fold cut in X traffic
// (the dominant HBM/L2 stream at 23.3 TB/s; weights are L2-resident).
// Irregular ops (FPS, kNN, gather, interp, maxpool, 128->2 head) are VALU kernels.
// ---------------------------------------------------------------------------

typedef __attribute__((ext_vector_type(16))) _Float16 v16h;
typedef __attribute__((ext_vector_type(8)))  _Float16 v8h;
typedef __attribute__((ext_vector_type(8)))  float    v8f;

#define BN_INV_SQRT 0.99950037462f   /* 1/sqrt(1 + 1e-3) */

// ---------------------------------------------------------------------------
// Weight packing: W[f32, Cin x Cout] * (gamma/sqrt(1+eps)) -> f16 in the exact
// WMMA B-fragment layout: index ((ct*kTiles + kt)*32 + lane)*16 + h, where the
// element is B[k = kt*32 + (h&7) + 16*(h>=8) + 8*(lane>=16)][n = ct*16 + (lane&15)].
// Rows k >= Cin are zero (Cin padded to a multiple of 32).
// ---------------------------------------------------------------------------
__global__ void pack_w_kernel(const float* __restrict__ W,
                              const float* __restrict__ gamma,
                              int Cin, int Cout, int CinPad,
                              _Float16* __restrict__ Wp) {
  int kTiles = CinPad >> 5;
  int colTiles = Cout >> 4;
  int total = colTiles * kTiles * 512;
  int t = blockIdx.x * blockDim.x + threadIdx.x;
  if (t >= total) return;
  int h    = t & 15;
  int lane = (t >> 4) & 31;
  int rest = t >> 9;
  int kt = rest % kTiles;
  int ct = rest / kTiles;
  int k = kt * 32 + (h & 7) + ((h >> 3) << 4) + (((lane >> 4) & 1) << 3);
  int n = ct * 16 + (lane & 15);
  float v = 0.f;
  if (k < Cin) {
    float s = gamma ? gamma[n] * BN_INV_SQRT : 1.0f;
    v = W[(size_t)k * Cout + n] * s;
  }
  Wp[t] = (_Float16)v;
}

// ---------------------------------------------------------------------------
// WMMA GEMM: Y[M,Cout](f16) = act( X[M,CinPad](f16) @ Wpacked + bias ).
// One wave32 per 16 x (NT*16) output strip; K stepped by 32.
// M % 16 == 0, CinPad % 32 == 0, Cout % (NT*16) == 0; full waves => EXEC all 1s.
// ---------------------------------------------------------------------------
template <int NT>
__global__ __launch_bounds__(128) void wmma_gemm_kernel(
    const _Float16* __restrict__ X, const _Float16* __restrict__ Wp,
    const float* __restrict__ bias, _Float16* __restrict__ Y,
    int M, int CinPad, int Cout, int relu) {
  int lane = threadIdx.x & 31;
  int wave = blockIdx.x * blockDim.y + threadIdx.y;
  int rowTiles = M >> 4;
  int colGroups = Cout >> 4 >> (NT == 4 ? 2 : 1);
  if (wave >= rowTiles * colGroups) return;
  int rt = wave % rowTiles;   // consecutive waves in a block share the B tiles
  int cg = wave / rowTiles;
  int kTiles = CinPad >> 5;
  int hiSel = (lane >> 4) & 1;

  // A fragment: lane holds row (rt*16 + lane&15); halves 0..7 = K base+0..7,
  // halves 8..15 = K base+16..23, with base += 8 for lanes 16..31.
  const _Float16* xrow = X + (size_t)((rt << 4) + (lane & 15)) * CinPad + (hiSel << 3);
  // B fragments: pre-packed, 16 contiguous halves per lane per k-tile; column
  // tile ct = cg*NT + j lives at offset ct * kTiles * 512.
  const size_t wstride = (size_t)kTiles * 512;
  const _Float16* wpp = Wp + (size_t)(cg * NT) * wstride + lane * 16;

  v8f acc[NT];
#pragma unroll
  for (int j = 0; j < NT; ++j) acc[j] = {};

  for (int kt = 0; kt < kTiles; ++kt) {
    v8h a0 = *(const v8h*)(xrow);
    v8h a1 = *(const v8h*)(xrow + 16);
    xrow += 32;
    __builtin_prefetch(xrow, 0, 1);   // global_prefetch_b8
    v16h A;
#pragma unroll
    for (int i = 0; i < 8; ++i) { A[i] = a0[i]; A[i + 8] = a1[i]; }

#pragma unroll
    for (int j = 0; j < NT; ++j) {
      const _Float16* wj = wpp + (size_t)j * wstride;
      v8h b0 = *(const v8h*)(wj);
      v8h b1 = *(const v8h*)(wj + 8);
      v16h Bm;
#pragma unroll
      for (int i = 0; i < 8; ++i) { Bm[i] = b0[i]; Bm[i + 8] = b1[i]; }
      acc[j] = __builtin_amdgcn_wmma_f32_16x16x32_f16(
          false, A, false, Bm, (short)0, acc[j], false, false);
    }
    wpp += 512;
  }

  // C/D layout: acc[j][v] is row rt*16 + v + 8*hiSel, col (cg*NT+j)*16+(lane&15).
  int rbase = (rt << 4) + (hiSel << 3);
#pragma unroll
  for (int j = 0; j < NT; ++j) {
    int n = ((cg * NT + j) << 4) + (lane & 15);
    float bv = bias ? bias[n] : 0.f;
#pragma unroll
    for (int v = 0; v < 8; ++v) {
      float y = acc[j][v] + bv;
      if (relu) y = fmaxf(y, 0.f);
      Y[(size_t)(rbase + v) * Cout + n] = (_Float16)y;
    }
  }
}

// ---------------------------------------------------------------------------
// Farthest point sampling: one block per batch, dmin[N] lives in LDS.
// ---------------------------------------------------------------------------
__global__ void fps_kernel(const float* __restrict__ xyz, int N, int npoint,
                           int* __restrict__ outIdx) {
  extern __shared__ float smem[];          // dmin[N] | rv[nt] | ri[nt]
  int b = blockIdx.x, tid = threadIdx.x, nt = blockDim.x;
  const float* p = xyz + (size_t)b * N * 3;
  float* dmin = smem;
  float* rv = smem + N;
  int*   ri = (int*)(rv + nt);
  for (int j = tid; j < N; j += nt) dmin[j] = 1e10f;
  __syncthreads();
  int far = 0;
  for (int i = 0; i < npoint; ++i) {
    if (tid == 0) outIdx[(size_t)b * npoint + i] = far;
    float cx = p[far * 3 + 0], cy = p[far * 3 + 1], cz = p[far * 3 + 2];
    float bestv = -1.f; int besti = 0;
    for (int j = tid; j < N; j += nt) {
      float dx = p[j * 3 + 0] - cx, dy = p[j * 3 + 1] - cy, dz = p[j * 3 + 2] - cz;
      float d = dx * dx + dy * dy + dz * dz;
      float m = fminf(dmin[j], d);
      dmin[j] = m;
      if (m > bestv) { bestv = m; besti = j; }
    }
    rv[tid] = bestv; ri[tid] = besti;
    __syncthreads();
    for (int s = nt >> 1; s > 0; s >>= 1) {
      if (tid < s) {
        if (rv[tid + s] > rv[tid]) { rv[tid] = rv[tid + s]; ri[tid] = ri[tid + s]; }
      }
      __syncthreads();
    }
    far = ri[0];
    __syncthreads();
  }
}

__global__ void gather_xyz_kernel(const float* __restrict__ xyz,
                                  const int* __restrict__ idx,
                                  float* __restrict__ out, int B, int S, int N) {
  int t = blockIdx.x * blockDim.x + threadIdx.x;
  if (t >= B * S) return;
  int b = t / S;
  int id = idx[t];
  const float* p = xyz + ((size_t)b * N + id) * 3;
  out[t * 3 + 0] = p[0]; out[t * 3 + 1] = p[1]; out[t * 3 + 2] = p[2];
}

// ---------------------------------------------------------------------------
// kNN: one thread per query, top-K by replace-the-worst.
// ---------------------------------------------------------------------------
template <int K, bool WD>
__global__ void knn_kernel(const float* __restrict__ q, const float* __restrict__ r,
                           int B, int S, int N,
                           int* __restrict__ idx, float* __restrict__ dout) {
  int t = blockIdx.x * blockDim.x + threadIdx.x;
  if (t >= B * S) return;
  int b = t / S;
  const float* Q = q + (size_t)t * 3;
  const float* R = r + (size_t)b * N * 3;
  float qx = Q[0], qy = Q[1], qz = Q[2];
  float kd[K]; int ki[K];
  for (int j = 0; j < K; ++j) { kd[j] = 3.4e38f; ki[j] = 0; }
  float worst = 3.4e38f; int wpos = 0;
  for (int n = 0; n < N; ++n) {
    float dx = R[n * 3 + 0] - qx, dy = R[n * 3 + 1] - qy, dz = R[n * 3 + 2] - qz;
    float d = dx * dx + dy * dy + dz * dz;
    if (d < worst) {
      kd[wpos] = d; ki[wpos] = n;
      worst = kd[0]; wpos = 0;
      for (int j = 1; j < K; ++j)
        if (kd[j] > worst) { worst = kd[j]; wpos = j; }
    }
  }
  size_t base = (size_t)t * K;
  for (int j = 0; j < K; ++j) {
    idx[base + j] = ki[j];
    if (WD) dout[base + j] = kd[j];
  }
}

// ---------------------------------------------------------------------------
// SA grouping: out[(b,s,k), 0:3] = xyz[idx]-new_xyz, then C gathered feats,
// zero-padded to CinPad (f16).
// ---------------------------------------------------------------------------
__global__ void sa_group_kernel(const float* __restrict__ xyz,
                                const float* __restrict__ nxyz,
                                const int* __restrict__ idx,
                                const _Float16* __restrict__ feats,
                                int B, int S, int K, int N, int C, int CinPad,
                                _Float16* __restrict__ out) {
  int t = blockIdx.x * blockDim.x + threadIdx.x;
  if (t >= B * S * K) return;
  int s = (t / K) % S;
  int b = t / (K * S);
  int id = idx[t];
  const float* pr = xyz + ((size_t)b * N + id) * 3;
  const float* pc = nxyz + ((size_t)b * S + s) * 3;
  _Float16* o = out + (size_t)t * CinPad;
  o[0] = (_Float16)(pr[0] - pc[0]);
  o[1] = (_Float16)(pr[1] - pc[1]);
  o[2] = (_Float16)(pr[2] - pc[2]);
  if (feats) {
    const _Float16* f = feats + ((size_t)b * N + id) * C;
    for (int c = 0; c < C; ++c) o[3 + c] = f[c];
  }
  for (int c = 3 + C; c < CinPad; ++c) o[c] = (_Float16)0.f;
}

__global__ void maxpool_kernel(const _Float16* __restrict__ X,
                               _Float16* __restrict__ Y,
                               int rows, int K, int C) {
  int t = blockIdx.x * blockDim.x + threadIdx.x;
  if (t >= rows * C) return;
  int c = t % C, r = t / C;
  const _Float16* p = X + (size_t)r * K * C + c;
  float m = (float)p[0];
  for (int k = 1; k < K; ++k) m = fmaxf(m, (float)p[(size_t)k * C]);
  Y[t] = (_Float16)m;
}

// ---------------------------------------------------------------------------
// FP grouping: 3-NN inverse-distance interp of f2, concat f1, pad to CinPad.
// ---------------------------------------------------------------------------
__global__ void fp_group_kernel(const int* __restrict__ idx,
                                const float* __restrict__ dist,
                                const _Float16* __restrict__ f2,
                                const _Float16* __restrict__ f1,
                                int B, int N1, int N2, int C2, int C1, int CinPad,
                                _Float16* __restrict__ out) {
  int t = blockIdx.x * blockDim.x + threadIdx.x;
  if (t >= B * N1) return;
  int b = t / N1;
  size_t kb = (size_t)t * 3;
  float w0 = 1.f / fmaxf(dist[kb + 0], 1e-10f);
  float w1 = 1.f / fmaxf(dist[kb + 1], 1e-10f);
  float w2 = 1.f / fmaxf(dist[kb + 2], 1e-10f);
  float ws = w0 + w1 + w2;
  w0 /= ws; w1 /= ws; w2 /= ws;
  const _Float16* r0 = f2 + ((size_t)b * N2 + idx[kb + 0]) * C2;
  const _Float16* r1 = f2 + ((size_t)b * N2 + idx[kb + 1]) * C2;
  const _Float16* r2 = f2 + ((size_t)b * N2 + idx[kb + 2]) * C2;
  _Float16* o = out + (size_t)t * CinPad;
  for (int c = 0; c < C2; ++c)
    o[c] = (_Float16)(w0 * (float)r0[c] + w1 * (float)r1[c] + w2 * (float)r2[c]);
  if (f1) {
    const _Float16* g = f1 + (size_t)t * C1;
    for (int c = 0; c < C1; ++c) o[C2 + c] = g[c];
  }
  for (int c = C2 + C1; c < CinPad; ++c) o[c] = (_Float16)0.f;
}

// Final head: out[B*N, 2] = h(f16,128) @ W2(f32,128x2) + b2 (f32 out).
__global__ void head_final_kernel(const _Float16* __restrict__ H,
                                  const float* __restrict__ W2,
                                  const float* __restrict__ b2,
                                  float* __restrict__ out, int rows) {
  int t = blockIdx.x * blockDim.x + threadIdx.x;
  if (t >= rows) return;
  const _Float16* h = H + (size_t)t * 128;
  float a0 = b2[0], a1 = b2[1];
  for (int c = 0; c < 128; ++c) {
    float x = (float)h[c];
    a0 += x * W2[c * 2 + 0];
    a1 += x * W2[c * 2 + 1];
  }
  out[t * 2 + 0] = a0;
  out[t * 2 + 1] = a1;
}

// ---------------------------------------------------------------------------
// Host orchestration.
// ---------------------------------------------------------------------------
extern "C" void kernel_launch(void* const* d_in, const int* in_sizes, int n_in,
                              void* d_out, int out_size, void* d_ws, size_t ws_size,
                              hipStream_t stream) {
  (void)in_sizes; (void)n_in; (void)out_size; (void)ws_size;
  const int B = 8;
  const float* x = (const float*)d_in[0];

  // Input-leaf mapping (dict/list/tuple flattening order of setup_inputs()):
  // 0:x ; sa1 1..9 ; sa2 10..18 ; sa3 19..27 ; sa4 28..36 ; fp4 37..42 ;
  // fp3 43..48 ; fp2 49..54 ; fp1 55..63 ; head 64:W1 65:b1 66:W2 67:b2
  // GEMM layers: {W idx, gamma idx (-1 = none), bias idx, Cin, Cout}
  static const int desc[22][5] = {
    { 1,  2,  3,   3,  32}, { 4,  5,  6,  32,  32}, { 7,  8,  9,  32,  64},   // sa1
    {10, 11, 12,  67,  64}, {13, 14, 15,  64,  64}, {16, 17, 18,  64, 128},   // sa2
    {19, 20, 21, 131, 128}, {22, 23, 24, 128, 128}, {25, 26, 27, 128, 256},   // sa3
    {28, 29, 30, 259, 256}, {31, 32, 33, 256, 256}, {34, 35, 36, 256, 512},   // sa4
    {37, 38, 39, 768, 256}, {40, 41, 42, 256, 256},                            // fp4
    {43, 44, 45, 384, 256}, {46, 47, 48, 256, 256},                            // fp3
    {49, 50, 51, 320, 256}, {52, 53, 54, 256, 128},                            // fp2
    {55, 56, 57, 128, 128}, {58, 59, 60, 128, 128}, {61, 62, 63, 128, 128},   // fp1
    {64, -1, 65, 128, 128},                                                    // head dense1
  };

  // Bump allocator over d_ws (256B aligned).
  char* ws = (char*)d_ws;
  size_t off = 0;
  auto alloc = [&](size_t bytes) -> void* {
    void* p = ws + off;
    off = (off + bytes + 255) & ~(size_t)255;
    return p;
  };

  // Pack all GEMM weights (BN gamma folded, f16, WMMA B layout).
  _Float16* wp[22];
  for (int i = 0; i < 22; ++i) {
    int Cin = desc[i][3], Cout = desc[i][4];
    int CinPad = (Cin + 31) & ~31;
    size_t nh = (size_t)CinPad * Cout;
    wp[i] = (_Float16*)alloc(nh * sizeof(_Float16));
    const float* gm = (desc[i][1] >= 0) ? (const float*)d_in[desc[i][1]] : nullptr;
    pack_w_kernel<<<((int)nh + 255) / 256, 256, 0, stream>>>(
        (const float*)d_in[desc[i][0]], gm, Cin, Cout, CinPad, wp[i]);
  }

  // Activation ping-pong buffers (max need: SA1, 262144 rows x 64ch f16 = 32MB).
  const size_t BUFH = (size_t)262144 * 64;
  _Float16* bufA = (_Float16*)alloc(BUFH * sizeof(_Float16));
  _Float16* bufB = (_Float16*)alloc(BUFH * sizeof(_Float16));

  float* l1x = (float*)alloc((size_t)B * 1024 * 3 * 4);
  float* l2x = (float*)alloc((size_t)B * 256 * 3 * 4);
  float* l3x = (float*)alloc((size_t)B * 64 * 3 * 4);
  float* l4x = (float*)alloc((size_t)B * 16 * 3 * 4);
  _Float16* l1f  = (_Float16*)alloc((size_t)B * 1024 * 64 * 2);
  _Float16* l2f  = (_Float16*)alloc((size_t)B * 256 * 128 * 2);
  _Float16* l3f  = (_Float16*)alloc((size_t)B * 64 * 256 * 2);
  _Float16* l4f  = (_Float16*)alloc((size_t)B * 16 * 512 * 2);
  _Float16* l3f2 = (_Float16*)alloc((size_t)B * 64 * 256 * 2);
  _Float16* l2f2 = (_Float16*)alloc((size_t)B * 256 * 256 * 2);
  _Float16* l1f2 = (_Float16*)alloc((size_t)B * 1024 * 128 * 2);
  _Float16* l0f  = (_Float16*)alloc((size_t)B * 8192 * 128 * 2);
  int*   fpsIdx = (int*)alloc((size_t)B * 1024 * 4);
  int*   knnIdx = (int*)alloc((size_t)B * 1024 * 32 * 4);
  int*   fpIdx  = (int*)alloc((size_t)B * 8192 * 3 * 4);
  float* fpDist = (float*)alloc((size_t)B * 8192 * 3 * 4);

  auto gemm = [&](const _Float16* X, int li, int M, _Float16* Y, int relu) {
    int Cin = desc[li][3], Cout = desc[li][4];
    int CinPad = (Cin + 31) & ~31;
    const float* bias = (const float*)d_in[desc[li][2]];
    int colTiles = Cout >> 4;
    dim3 blk(32, 4);
    if ((colTiles & 3) == 0) {
      int waves = (M >> 4) * (colTiles >> 2);
      wmma_gemm_kernel<4><<<(waves + 3) / 4, blk, 0, stream>>>(
          X, wp[li], bias, Y, M, CinPad, Cout, relu);
    } else {
      int waves = (M >> 4) * (colTiles >> 1);
      wmma_gemm_kernel<2><<<(waves + 3) / 4, blk, 0, stream>>>(
          X, wp[li], bias, Y, M, CinPad, Cout, relu);
    }
  };

  auto sa_stage = [&](const float* xyzIn, int N, const _Float16* featsIn, int C,
                      float* xyzOut, int S, _Float16* featsOut, int l0) {
    size_t shmem = (size_t)N * 4 + 256 * 8;
    fps_kernel<<<B, 256, shmem, stream>>>(xyzIn, N, S, fpsIdx);
    gather_xyz_kernel<<<(B * S + 255) / 256, 256, 0, stream>>>(
        xyzIn, fpsIdx, xyzOut, B, S, N);
    knn_kernel<32, false><<<(B * S + 127) / 128, 128, 0, stream>>>(
        xyzOut, xyzIn, B, S, N, knnIdx, nullptr);
    int CinPad = (3 + C + 31) & ~31;
    sa_group_kernel<<<(B * S * 32 + 255) / 256, 256, 0, stream>>>(
        xyzIn, xyzOut, knnIdx, featsIn, B, S, 32, N, C, CinPad, bufA);
    int M = B * S * 32;
    gemm(bufA, l0,     M, bufB, 1);
    gemm(bufB, l0 + 1, M, bufA, 1);
    gemm(bufA, l0 + 2, M, bufB, 1);
    int Cout = desc[l0 + 2][4];
    maxpool_kernel<<<(B * S * Cout + 255) / 256, 256, 0, stream>>>(
        bufB, featsOut, B * S, 32, Cout);
  };

  auto fp_stage = [&](const float* xyz1, int N1, const float* xyz2, int N2,
                      const _Float16* f1, int C1, const _Float16* f2, int C2,
                      _Float16* outF, int l0, int nLayers) {
    knn_kernel<3, true><<<(B * N1 + 127) / 128, 128, 0, stream>>>(
        xyz1, xyz2, B, N1, N2, fpIdx, fpDist);
    int CinPad = (C2 + C1 + 31) & ~31;
    fp_group_kernel<<<(B * N1 + 255) / 256, 256, 0, stream>>>(
        fpIdx, fpDist, f2, f1, B, N1, N2, C2, C1, CinPad, bufA);
    int M = B * N1;
    if (nLayers == 2) {
      gemm(bufA, l0,     M, bufB, 1);
      gemm(bufB, l0 + 1, M, outF, 1);
    } else {  // 3 layers (fp1)
      gemm(bufA, l0,     M, bufB, 1);
      gemm(bufB, l0 + 1, M, bufA, 1);
      gemm(bufA, l0 + 2, M, outF, 1);
    }
  };

  // Encoder (set abstraction).
  sa_stage(x,   8192, nullptr, 0,   l1x, 1024, l1f, 0);
  sa_stage(l1x, 1024, l1f,     64,  l2x, 256,  l2f, 3);
  sa_stage(l2x, 256,  l2f,     128, l3x, 64,   l3f, 6);
  sa_stage(l3x, 64,   l3f,     256, l4x, 16,   l4f, 9);

  // Decoder (feature propagation).
  fp_stage(l3x, 64,   l4x, 16,   l3f, 256, l4f,  512, l3f2, 12, 2);
  fp_stage(l2x, 256,  l3x, 64,   l2f, 128, l3f2, 256, l2f2, 14, 2);
  fp_stage(l1x, 1024, l2x, 256,  l1f, 64,  l2f2, 256, l1f2, 16, 2);
  fp_stage(x,   8192, l1x, 1024, nullptr, 0, l1f2, 128, l0f, 18, 3);

  // Head: Dense(128)+ReLU via WMMA, then tiny 128->2 dense in f32.
  gemm(l0f, 21, B * 8192, bufA, 1);
  head_final_kernel<<<(B * 8192 + 255) / 256, 256, 0, stream>>>(
      bufA, (const float*)d_in[66], (const float*)d_in[67], (float*)d_out, B * 8192);
}
